// LSNet_1D_46042049413174
// MI455X (gfx1250) — matrix operand order repaired and checked by reading, not compiled, not run on hardware
//
#include <hip/hip_runtime.h>
#include <hip/hip_bf16.h>
#include <math.h>

typedef _Float16 v16h __attribute__((ext_vector_type(16)));
typedef float    v8f  __attribute__((ext_vector_type(8)));

#define CHg 128
#define Lg  2048
#define Bg  2
#define Ng  (Bg*Lg)   // 4096 positions, channel-major layout (C, B*L)

static __device__ __forceinline__ int iabs_(int a) { return a < 0 ? -a : a; }

// ---- CDNA5 async global->LDS copy (16B per lane), tracked by ASYNCcnt ----------
static __device__ __forceinline__ void async_g2lds_b128(unsigned lds_off, const void* g) {
  asm volatile("global_load_async_to_lds_b128 %0, %1, off"
               :: "v"(lds_off), "v"(g) : "memory");
}
static __device__ __forceinline__ void wait_asynccnt0() {
  asm volatile("s_wait_asynccnt 0x0" ::: "memory");
}

// ---------------- stem: conv1d(1->128, K=5, pad=2) + bias + relu ----------------
__global__ void stem_kernel(const float* __restrict__ x, const float* __restrict__ W,
                            const float* __restrict__ bias, float* __restrict__ h) {
  int i = blockIdx.x * 256 + threadIdx.x;
  if (i >= CHg * Ng) return;
  int c = i / Ng, n = i % Ng, b = n >> 11, l = n & (Lg - 1);
  float s = bias[c];
#pragma unroll
  for (int k = 0; k < 5; ++k) {
    int ll = l + k - 2;
    if (ll >= 0 && ll < Lg) s += x[b * Lg + ll] * W[c * 5 + k];
  }
  h[i] = fmaxf(s, 0.f);
}

// ---------------- WMMA GEMM: Y[M,Ng] = A[M,K] @ X[K,Ng]  (f16 in, f32 acc) ------
// grid = (M/16, Ng/64), block = 128 (4 waves). X tiles (32K x 64N f32) staged to
// LDS via async copies, double-buffered; weights (small, L2-resident) from global.
__global__ void __launch_bounds__(128) gemm_wmma_kernel(
    const float* __restrict__ A, const float* __restrict__ X,
    float* __restrict__ Y, int M, int K) {
  __shared__ float xtile[2][32 * 64];  // 2 x 8KB
  const int Nn = Ng;
  int wave = threadIdx.x >> 5;
  int lane = threadIdx.x & 31;
  int g = lane >> 4, lm = lane & 15;
  int M0 = blockIdx.x * 16;
  int N0blk = blockIdx.y * 64;

  auto issueX = [&](int k0, int buf) {
    // 32 rows x 64 cols f32 = 512 segments of 16B; 4 per thread
#pragma unroll
    for (int rep = 0; rep < 4; ++rep) {
      int j = rep * 128 + threadIdx.x;
      int rr = j >> 4, seg = j & 15;
      const float* gp = X + (size_t)(k0 + rr) * Nn + N0blk + seg * 4;
      unsigned lds = (unsigned)(uintptr_t)&xtile[buf][rr * 64 + seg * 4];
      async_g2lds_b128(lds, gp);
    }
  };

  v8f acc = {0.f, 0.f, 0.f, 0.f, 0.f, 0.f, 0.f, 0.f};
  issueX(0, 0);
  int nsteps = K / 32;
  for (int it = 0; it < nsteps; ++it) {
    int k0 = it * 32, buf = it & 1;
    wait_asynccnt0();
    __syncthreads();                       // tile[buf] ready; tile[buf^1] free
    if (it + 1 < nsteps) issueX(k0 + 32, buf ^ 1);
    v16h a, b;
#pragma unroll
    for (int e = 0; e < 16; ++e) {  // A layout: M=lane%16, K halves by lane group
      int kk = k0 + e + 16 * g;
      a[e] = (_Float16)A[(M0 + lm) * K + kk];
    }
#pragma unroll
    for (int e = 0; e < 16; ++e) {  // B layout: N=lane%16, K = e + 16*g
      int kk = e + 16 * g;
      b[e] = (_Float16)xtile[buf][kk * 64 + wave * 16 + lm];
    }
    acc = __builtin_amdgcn_wmma_f32_16x16x32_f16(false, a, false, b, (short)0, acc,
                                                 false, false);
  }
#pragma unroll
  for (int r = 0; r < 8; ++r)  // C layout: M = r + 8*(lane/16), N = lane%16
    Y[(M0 + r + 8 * g) * Nn + N0blk + wave * 16 + lm] = acc[r];
}

// ---------------- per-channel batch stats (mean, biased var over Ng) ------------
__global__ void stats_kernel(const float* __restrict__ x, float* __restrict__ st) {
  int c = blockIdx.x;
  __shared__ float ss[256], sq[256];
  float s = 0.f, q = 0.f;
  for (int i = threadIdx.x; i < Ng; i += 256) {
    float v = x[c * Ng + i];
    s += v; q += v * v;
  }
  ss[threadIdx.x] = s; sq[threadIdx.x] = q;
  __syncthreads();
  for (int o = 128; o > 0; o >>= 1) {
    if (threadIdx.x < o) { ss[threadIdx.x] += ss[threadIdx.x + o]; sq[threadIdx.x] += sq[threadIdx.x + o]; }
    __syncthreads();
  }
  if (threadIdx.x == 0) {
    float m = ss[0] / (float)Ng;
    st[2 * c] = m;
    st[2 * c + 1] = sq[0] / (float)Ng - m * m;
  }
}

// ---------------- BN affine + activation + optional residual --------------------
// act: 0=none, 1=relu, 2=silu
__global__ void bn_apply_kernel(const float* __restrict__ x, const float* __restrict__ st,
                                const float* __restrict__ gw, const float* __restrict__ bw,
                                const float* __restrict__ resid, float* __restrict__ y,
                                int C, int act) {
  int i = blockIdx.x * 256 + threadIdx.x;
  if (i >= C * Ng) return;
  int c = i / Ng;
  float m = st[2 * c], v = st[2 * c + 1];
  float val = (x[i] - m) * rsqrtf(v + 1e-5f) * gw[c] + bw[c];
  if (act == 1) val = fmaxf(val, 0.f);
  else if (act == 2) val = val / (1.f + __expf(-val));
  if (resid) val += resid[i];
  y[i] = val;
}

// ---------------- GroupNorm over (b, group): 16 groups x 3 ch x L ---------------
__global__ void gnorm_stats_kernel(const float* __restrict__ x, const float* __restrict__ bias,
                                   float* __restrict__ st) {
  int b = blockIdx.x >> 4, grp = blockIdx.x & 15;
  __shared__ float ss[256], sq[256];
  float s = 0.f, q = 0.f;
  for (int i = threadIdx.x; i < 3 * Lg; i += 256) {
    int j = i / Lg, l = i % Lg;
    int ch = grp * 3 + j;
    float v = x[ch * Ng + b * Lg + l] + bias[ch];
    s += v; q += v * v;
  }
  ss[threadIdx.x] = s; sq[threadIdx.x] = q;
  __syncthreads();
  for (int o = 128; o > 0; o >>= 1) {
    if (threadIdx.x < o) { ss[threadIdx.x] += ss[threadIdx.x + o]; sq[threadIdx.x] += sq[threadIdx.x + o]; }
    __syncthreads();
  }
  if (threadIdx.x == 0) {
    float m = ss[0] / (float)(3 * Lg);
    st[2 * blockIdx.x] = m;
    st[2 * blockIdx.x + 1] = sq[0] / (float)(3 * Lg) - m * m;
  }
}

__global__ void gnorm_apply_kernel(const float* __restrict__ x, const float* __restrict__ bias,
                                   const float* __restrict__ st, const float* __restrict__ gg,
                                   const float* __restrict__ gb, float* __restrict__ y) {
  int i = blockIdx.x * 256 + threadIdx.x;
  if (i >= 48 * Ng) return;
  int ch = i / Ng, n = i % Ng, b = n >> 11, grp = ch / 3;
  int si = b * 16 + grp;
  float m = st[2 * si], v = st[2 * si + 1];
  y[i] = (x[i] + bias[ch] - m) * rsqrtf(v + 1e-5f) * gg[ch] + gb[ch];
}

// ---------------- depthwise conv along L (per-channel kernel) -------------------
__global__ void dwconv_kernel(const float* __restrict__ x, const float* __restrict__ W,
                              float* __restrict__ y, int C, int Kk) {
  int i = blockIdx.x * 256 + threadIdx.x;
  if (i >= C * Ng) return;
  int c = i / Ng, n = i % Ng, l = n & (Lg - 1);
  int pad = (Kk - 1) >> 1;
  float s = 0.f;
  for (int k = 0; k < Kk; ++k) {
    int ll = l + k - pad;
    if (ll >= 0 && ll < Lg) s += x[i + (ll - l)] * W[c * Kk + k];
  }
  y[i] = s;
}

// ---------------- SKA: position-dependent K=3 kernel from w48 -------------------
__global__ void ska_kernel(const float* __restrict__ h, const float* __restrict__ w48,
                           float* __restrict__ y) {
  int i = blockIdx.x * 256 + threadIdx.x;
  if (i >= CHg * Ng) return;
  int c = i / Ng, n = i % Ng, l = n & (Lg - 1);
  float s = 0.f;
#pragma unroll
  for (int k = 0; k < 3; ++k) {
    int ll = l + k - 1;
    if (ll >= 0 && ll < Lg) s += h[i + (ll - l)] * w48[((c & 15) * 3 + k) * Ng + n];
  }
  y[i] = s;
}

// ---------------- fused flash attention (WMMA 16x16x32 f16) ---------------------
// grid = B*H*(L/128) = 256 blocks, 256 threads = 8 waves; each wave: 16 queries.
// K/V tiles (16+64 rows x 32 keys, f32) shared by all 8 waves, staged to LDS via
// async copies with double buffering. P tiles bounce through per-wave LDS for the
// C-layout -> A-layout transpose. ReLU fused on the O store.
__global__ void __launch_bounds__(256) attn_kernel(
    const float* __restrict__ q, const float* __restrict__ qkv,
    const float* __restrict__ ab, float* __restrict__ o) {
  __shared__ _Float16 plds[8][16 * 32];   // 8KB
  __shared__ float kvlds[2][80 * 32];     // 2 x 10KB: rows 0..15 = K, 16..79 = V
  int wave = threadIdx.x >> 5, lane = threadIdx.x & 31;
  int g = lane >> 4, lm = lane & 15;
  int qb = blockIdx.x & 15;
  int bh = blockIdx.x >> 4;
  int hh = bh & 7, bb = bh >> 3;
  int n0 = qb * 128 + wave * 16;
  const int Nn = Ng;

  auto issueKV = [&](int m0, int buf) {
    // (16 K-rows + 64 V-rows) x 32 cols f32 = 640 segments of 16B
    for (int j = threadIdx.x; j < 640; j += 256) {
      int rr = j >> 3, seg = j & 7;
      int grow = (rr < 16) ? (128 + hh * 16 + rr) : (256 + hh * 64 + (rr - 16));
      const float* gp = qkv + (size_t)grow * Nn + bb * Lg + m0 + seg * 4;
      unsigned lds = (unsigned)(uintptr_t)&kvlds[buf][rr * 32 + seg * 4];
      async_g2lds_b128(lds, gp);
    }
  };

  // A = Q tile (16 queries x K=32, d padded 16..31 with zeros)
  v16h aq;
#pragma unroll
  for (int e = 0; e < 16; ++e) {
    int d = e + 16 * g;
    aq[e] = (d < 16) ? (_Float16)q[(hh * 16 + d) * Nn + bb * Lg + n0 + lm]
                     : (_Float16)0.f;
  }

  v8f oacc[4];
#pragma unroll
  for (int t = 0; t < 4; ++t) oacc[t] = (v8f){0.f, 0.f, 0.f, 0.f, 0.f, 0.f, 0.f, 0.f};
  float mrun[8], lrun[8];
#pragma unroll
  for (int r = 0; r < 8; ++r) { mrun[r] = -1e30f; lrun[r] = 0.f; }

  issueKV(0, 0);
  for (int m0 = 0; m0 < Lg; m0 += 32) {
    int buf = (m0 >> 5) & 1;
    wait_asynccnt0();
    __syncthreads();                        // kv[buf] ready; kv[buf^1] free
    if (m0 + 32 < Lg) issueKV(m0 + 32, buf ^ 1);

    v8f s01[2];
#pragma unroll
    for (int hb = 0; hb < 2; ++hb) {  // two 16-key QK tiles
      v16h bk;
#pragma unroll
      for (int e = 0; e < 16; ++e) {
        int d = e + 16 * g;
        bk[e] = (d < 16) ? (_Float16)kvlds[buf][d * 32 + hb * 16 + lm]
                         : (_Float16)0.f;
      }
      v8f z = {0.f, 0.f, 0.f, 0.f, 0.f, 0.f, 0.f, 0.f};
      s01[hb] = __builtin_amdgcn_wmma_f32_16x16x32_f16(false, aq, false, bk, (short)0, z,
                                                       false, false);
    }
    // online softmax: row n = r + 8*g spread across 16 lanes (col = lane%16)
#pragma unroll
    for (int r = 0; r < 8; ++r) {
      int nq = n0 + r + 8 * g;
      float v0 = s01[0][r] * 0.25f + ab[hh * Lg + iabs_(nq - (m0 + lm))];
      float v1 = s01[1][r] * 0.25f + ab[hh * Lg + iabs_(nq - (m0 + 16 + lm))];
      float mx = fmaxf(v0, v1);
#pragma unroll
      for (int off = 1; off < 16; off <<= 1) mx = fmaxf(mx, __shfl_xor(mx, off, 32));
      float mnew = fmaxf(mrun[r], mx);
      float sc = __expf(mrun[r] - mnew);
      float p0 = __expf(v0 - mnew);
      float p1 = __expf(v1 - mnew);
      float rs = p0 + p1;
#pragma unroll
      for (int off = 1; off < 16; off <<= 1) rs += __shfl_xor(rs, off, 32);
      lrun[r] = lrun[r] * sc + rs;
      mrun[r] = mnew;
#pragma unroll
      for (int t = 0; t < 4; ++t) oacc[t][r] *= sc;
      plds[wave][(r + 8 * g) * 32 + lm]      = (_Float16)p0;  // C-layout -> LDS
      plds[wave][(r + 8 * g) * 32 + 16 + lm] = (_Float16)p1;
    }
    __syncthreads();
    // read P back in A-layout (rows = queries, K = 32 keys)
    v16h ap;
#pragma unroll
    for (int e = 0; e < 16; ++e) {
      int mm = e + 16 * g;
      ap[e] = plds[wave][lm * 32 + mm];
    }
    // O += P @ V^T : 4 tiles of d' (64 total), full K=32 depth
#pragma unroll
    for (int t = 0; t < 4; ++t) {
      v16h bv;
#pragma unroll
      for (int e = 0; e < 16; ++e)
        bv[e] = (_Float16)kvlds[buf][(16 + t * 16 + lm) * 32 + e + 16 * g];
      oacc[t] = __builtin_amdgcn_wmma_f32_16x16x32_f16(false, ap, false, bv, (short)0,
                                                       oacc[t], false, false);
    }
  }
#pragma unroll
  for (int r = 0; r < 8; ++r) {
    float inv = 1.f / lrun[r];
#pragma unroll
    for (int t = 0; t < 4; ++t) {
      float val = oacc[t][r] * inv;  // C-layout: row n = r+8g (query), col = d'
      o[(hh * 64 + t * 16 + lm) * Nn + bb * Lg + n0 + r + 8 * g] = fmaxf(val, 0.f);
    }
  }
}

// ---------------- final layout change (C,B*L) -> (B,C,L) ------------------------
__global__ void out_kernel(const float* __restrict__ h, float* __restrict__ out) {
  int i = blockIdx.x * 256 + threadIdx.x;
  if (i >= Bg * CHg * Lg) return;
  int b = i / (CHg * Lg), c = (i / Lg) % CHg, l = i % Lg;
  out[i] = h[c * Ng + b * Lg + l];
}

extern "C" void kernel_launch(void* const* d_in, const int* in_sizes, int n_in,
                              void* d_out, int out_size, void* d_ws, size_t ws_size,
                              hipStream_t stream) {
  (void)in_sizes; (void)n_in; (void)out_size; (void)ws_size;
  const float* x      = (const float*)d_in[0];
  const float* Wstem  = (const float*)d_in[1];
  const float* bstem  = (const float*)d_in[2];
  const float* cv1W   = (const float*)d_in[3];
  const float* cv1g   = (const float*)d_in[4];
  const float* cv1b   = (const float*)d_in[5];
  const float* cv2W   = (const float*)d_in[6];
  const float* cv2g   = (const float*)d_in[7];
  const float* cv2b   = (const float*)d_in[8];
  const float* cv3W   = (const float*)d_in[9];
  const float* cv3g   = (const float*)d_in[10];
  const float* cv3b   = (const float*)d_in[11];
  const float* cv4W   = (const float*)d_in[12];
  const float* cv4b   = (const float*)d_in[13];
  const float* gng    = (const float*)d_in[14];
  const float* gnb    = (const float*)d_in[15];
  const float* lsg    = (const float*)d_in[16];
  const float* lsb    = (const float*)d_in[17];
  const float* qkvW   = (const float*)d_in[18];
  const float* qkvg   = (const float*)d_in[19];
  const float* qkvb   = (const float*)d_in[20];
  const float* dwW    = (const float*)d_in[21];
  const float* dwg    = (const float*)d_in[22];
  const float* dwb    = (const float*)d_in[23];
  const float* projW  = (const float*)d_in[24];
  const float* projg  = (const float*)d_in[25];
  const float* projb  = (const float*)d_in[26];
  const float* ab     = (const float*)d_in[27];
  const float* pw1W   = (const float*)d_in[28];
  const float* pw1g   = (const float*)d_in[29];
  const float* pw1b   = (const float*)d_in[30];
  const float* pw2W   = (const float*)d_in[31];
  const float* pw2g   = (const float*)d_in[32];
  const float* pw2b   = (const float*)d_in[33];

  float* ws   = (float*)d_ws;
  float* h    = ws;  ws += CHg * Ng;   // 128 x 4096
  float* t1   = ws;  ws += 64  * Ng;
  float* t2   = ws;  ws += 64  * Ng;
  float* w48  = ws;  ws += 48  * Ng;
  float* skb  = ws;  ws += CHg * Ng;
  float* qkvB = ws;  ws += 768 * Ng;
  float* qdw  = ws;  ws += CHg * Ng;
  float* oB   = ws;  ws += 512 * Ng;
  float* ffn  = ws;  ws += 256 * Ng;
  float* raw  = ws;  ws += 768 * Ng;   // pre-norm GEMM/conv output
  float* st   = ws;  ws += 2 * 768;    // BN stats
  float* gst  = ws;  ws += 2 * 32;     // GN stats

  auto cdiv = [](int a, int b) { return (a + b - 1) / b; };
  auto gemm = [&](const float* A, const float* X_, float* Y, int M, int K) {
    dim3 grid(M / 16, Ng / 64);
    gemm_wmma_kernel<<<grid, 128, 0, stream>>>(A, X_, Y, M, K);
  };
  auto stats = [&](const float* X_, int C) {
    stats_kernel<<<C, 256, 0, stream>>>(X_, st);
  };
  auto apply = [&](const float* X_, const float* gw, const float* bw,
                   const float* resid, float* Y, int C, int act) {
    bn_apply_kernel<<<cdiv(C * Ng, 256), 256, 0, stream>>>(X_, st, gw, bw, resid, Y, C, act);
  };

  // stem
  stem_kernel<<<cdiv(CHg * Ng, 256), 256, 0, stream>>>(x, Wstem, bstem, h);
  // cv1: 128->64, BN, silu
  gemm(cv1W, h, raw, 64, 128); stats(raw, 64); apply(raw, cv1g, cv1b, nullptr, t1, 64, 2);
  // cv2: depthwise K=7, BN
  dwconv_kernel<<<cdiv(64 * Ng, 256), 256, 0, stream>>>(t1, cv2W, raw, 64, 7);
  stats(raw, 64); apply(raw, cv2g, cv2b, nullptr, t2, 64, 0);
  // cv3: 64->64, BN, silu
  gemm(cv3W, t2, raw, 64, 64); stats(raw, 64); apply(raw, cv3g, cv3b, nullptr, t1, 64, 2);
  // cv4: 64->48 (+bias) + GroupNorm(16 groups)
  gemm(cv4W, t1, raw, 48, 64);
  gnorm_stats_kernel<<<32, 256, 0, stream>>>(raw, cv4b, gst);
  gnorm_apply_kernel<<<cdiv(48 * Ng, 256), 256, 0, stream>>>(raw, cv4b, gst, gng, gnb, w48);
  // SKA + BN + residual -> h
  ska_kernel<<<cdiv(CHg * Ng, 256), 256, 0, stream>>>(h, w48, skb);
  stats(skb, 128); apply(skb, lsg, lsb, h, h, 128, 0);
  // qkv: 128->768, BN
  gemm(qkvW, h, raw, 768, 128); stats(raw, 768); apply(raw, qkvg, qkvb, nullptr, qkvB, 768, 0);
  // q depthwise K=3 + BN (rows 0..127 of qkv are q)
  dwconv_kernel<<<cdiv(128 * Ng, 256), 256, 0, stream>>>(qkvB, dwW, raw, 128, 3);
  stats(raw, 128); apply(raw, dwg, dwb, nullptr, qdw, 128, 0);
  // fused flash attention (relu fused on O store)
  attn_kernel<<<Bg * 8 * (Lg / 128), 256, 0, stream>>>(qdw, qkvB, ab, oB);
  // proj: 512->128, BN, residual -> h
  gemm(projW, oB, raw, 128, 512); stats(raw, 128); apply(raw, projg, projb, h, h, 128, 0);
  // FFN: pw1 (BN+relu), pw2 (BN+residual)
  gemm(pw1W, h, raw, 256, 128); stats(raw, 256); apply(raw, pw1g, pw1b, nullptr, ffn, 256, 1);
  gemm(pw2W, ffn, raw, 128, 256); stats(raw, 128); apply(raw, pw2g, pw2b, h, h, 128, 0);
  // (C, B*L) -> (B, C, L)
  out_kernel<<<cdiv(Bg * CHg * Lg, 256), 256, 0, stream>>>(h, (float*)d_out);
}
